// DynamicPropertyBank_18184891531793
// MI455X (gfx1250) — matrix operand order repaired and checked by reading, not compiled, not run on hardware
//
#include <hip/hip_runtime.h>
#include <hip/hip_bf16.h>
#include <math.h>

// ---------------- problem constants ----------------
#define BATCH      16384
#define INPUT_DIM  256
#define NUM_SLOTS  32
#define SLOT_DIM   64
#define NUM_KNOWN  9
#define HIDDEN     128
#define ITERS      3
#define LN_EPS     1e-5f

typedef __attribute__((ext_vector_type(16))) _Float16     v16h;
typedef __attribute__((ext_vector_type(2)))  _Float16     h2;
typedef __attribute__((ext_vector_type(8)))  float        v8f;
typedef __attribute__((ext_vector_type(8)))  unsigned int v8u;

// ---------------- WMMA helpers ----------------
__device__ __forceinline__ v8f wmma16(v16h a, v16h b, v8f c) {
  // v_wmma_f32_16x16x32_f16 : D = A(16x32 f16) x B(32x16 f16) + C(16x16 f32)
  return __builtin_amdgcn_wmma_f32_16x16x32_f16(false, a, false, b, (short)0, c,
                                                false, false);
}

// Build an A fragment (16x32, f16) from an f32 row-major LDS tile.
// ISA layout: lanes 0-15 = rows, g=lane>>4 selects K interleave:
//   halves 2p,2p+1 <- K = 8*g + (p<4 ? 2p : 16+2(p-4)) .. +1   (pairs contiguous)
__device__ __forceinline__ v16h ldsA_f32(const float* base, int stride, int kbase, int lane) {
  const int g = (lane >> 4) & 1, m = lane & 15;
  v8u u;
#pragma unroll
  for (int p = 0; p < 8; ++p) {
    const int k0 = 8 * g + (p < 4 ? 2 * p : 16 + 2 * (p - 4));
    const float* s = base + m * stride + kbase + k0;
    h2 hv; hv[0] = (_Float16)s[0]; hv[1] = (_Float16)s[1];
    u[p] = __builtin_bit_cast(unsigned int, hv);
  }
  return __builtin_bit_cast(v16h, u);
}

// Same but source is already f16 in LDS (4-byte aligned pair reads).
__device__ __forceinline__ v16h ldsA_f16(const _Float16* base, int stride, int kbase, int lane) {
  const int g = (lane >> 4) & 1, m = lane & 15;
  v8u u;
#pragma unroll
  for (int p = 0; p < 8; ++p) {
    const int k0 = 8 * g + (p < 4 ? 2 * p : 16 + 2 * (p - 4));
    u[p] = *(const unsigned int*)(base + m * stride + kbase + k0);
  }
  return __builtin_bit_cast(v16h, u);
}

// B fragment: pre-swizzled in global memory, 512 halves per fragment,
// lane l reads its 16 contiguous halves (32B -> 2x global_load_b128).
__device__ __forceinline__ v16h ldB(const _Float16* w, int frag, int lane) {
  return *(const v16h*)(w + (size_t)frag * 512 + lane * 16);
}

__device__ __forceinline__ float sigm(float x) { return 1.f / (1.f + expf(-x)); }
__device__ __forceinline__ float gelu_exact(float x) {
  return 0.5f * x * (1.f + erff(x * 0.70710678118654752f));
}

// ---------------- kernel 0: weight f32 -> f16 fragment swizzle ----------------
// B-matrix layout (16x16x32): fragment f = nt*KT + kt ; within fragment:
// lane l: n = nt*16 + (l&15), g = l>>4; half j -> k = kt*32 + (j<8 ? 8g + j : 16 + 8g + (j-8)).
__global__ void swizzle_w(const float* __restrict__ W, _Float16* __restrict__ out,
                          int K, int N) {
  int i = blockIdx.x * 256 + threadIdx.x;
  int total = K * N;
  if (i >= total) return;
  int f  = i >> 9;
  int l  = (i >> 4) & 31;
  int j  = i & 15;
  int KT = K >> 5;
  int kt = f % KT, nt = f / KT;
  int g  = l >> 4;
  int n  = nt * 16 + (l & 15);
  int k  = kt * 32 + (j < 8 ? 8 * g + j : 16 + 8 * g + (j - 8));
  out[i] = (_Float16)W[k * N + n];
}

// ---------------- kernel 1: preproc (x = LN(ws@Wi+bi); v = x@Wv+bv; gi = v@W_ih+b_ih)
__global__ __launch_bounds__(128) void preproc(
    const float* __restrict__ ws, const _Float16* __restrict__ wi,
    const float* __restrict__ bi, const _Float16* __restrict__ wv,
    const float* __restrict__ bv, const _Float16* __restrict__ wih,
    const float* __restrict__ bih, const float* __restrict__ g_in,
    const float* __restrict__ b_in, float* __restrict__ gi_g) {
  __shared__ __align__(16) _Float16 in16[64 * 256];  // 32 KB
  __shared__ __align__(16) float    act[64 * 64];    // 16 KB (x, then v)

  const int tid = threadIdx.x;
  const int wave = tid >> 5, lane = tid & 31;
  const int g = lane >> 4, ln15 = lane & 15;
  const size_t r0 = (size_t)blockIdx.x * 64;
  const int rowBase = wave * 16;

  for (int i = tid; i < 64 * 256; i += 128)
    in16[i] = (_Float16)ws[r0 * 256 + i];
  __syncthreads();

  // x = ws @ Wi + bi   (K=256: 8 k-steps, N=64: 4 n-tiles)
  v16h av[8];
#pragma unroll
  for (int kt = 0; kt < 8; ++kt)
    av[kt] = ldsA_f16(in16 + rowBase * 256, 256, kt * 32, lane);
#pragma unroll
  for (int nt = 0; nt < 4; ++nt) {
    v8f c = {};
#pragma unroll
    for (int kt = 0; kt < 8; ++kt) c = wmma16(av[kt], ldB(wi, nt * 8 + kt, lane), c);
    int col = nt * 16 + ln15;
    float bic = bi[col];
#pragma unroll
    for (int r8 = 0; r8 < 8; ++r8)
      act[(rowBase + r8 + 8 * g) * 64 + col] = c[r8] + bic;
  }
  __syncthreads();

  // LayerNorm per row (64 rows, one thread each)
  if (tid < 64) {
    float* rp = act + tid * 64;
    float m = 0.f;
    for (int d = 0; d < 64; ++d) m += rp[d];
    m *= (1.f / 64.f);
    float var = 0.f;
    for (int d = 0; d < 64; ++d) { float t = rp[d] - m; var += t * t; }
    var *= (1.f / 64.f);
    float inv = rsqrtf(var + LN_EPS);
    for (int d = 0; d < 64; ++d) rp[d] = (rp[d] - m) * inv * g_in[d] + b_in[d];
  }
  __syncthreads();

  // v = x @ Wv + bv
  v16h xa0 = ldsA_f32(act + rowBase * 64, 64, 0, lane);
  v16h xa1 = ldsA_f32(act + rowBase * 64, 64, 32, lane);
  v8f vacc[4];
#pragma unroll
  for (int nt = 0; nt < 4; ++nt) {
    v8f c = {};
    c = wmma16(xa0, ldB(wv, nt * 2 + 0, lane), c);
    c = wmma16(xa1, ldB(wv, nt * 2 + 1, lane), c);
    vacc[nt] = c;
  }
  __syncthreads();  // everyone done reading x
#pragma unroll
  for (int nt = 0; nt < 4; ++nt) {
    int col = nt * 16 + ln15;
    float bvc = bv[col];
#pragma unroll
    for (int r8 = 0; r8 < 8; ++r8)
      act[(rowBase + r8 + 8 * g) * 64 + col] = vacc[nt][r8] + bvc;
  }
  __syncthreads();

  // gi = v @ W_ih + b_ih  (N=192: 12 n-tiles) -> global workspace (f32)
  v16h va0 = ldsA_f32(act + rowBase * 64, 64, 0, lane);
  v16h va1 = ldsA_f32(act + rowBase * 64, 64, 32, lane);
#pragma unroll
  for (int nt = 0; nt < 12; ++nt) {
    v8f c = {};
    c = wmma16(va0, ldB(wih, nt * 2 + 0, lane), c);
    c = wmma16(va1, ldB(wih, nt * 2 + 1, lane), c);
    int col = nt * 16 + ln15;
    float bc = bih[col];
#pragma unroll
    for (int r8 = 0; r8 < 8; ++r8) {
      size_t brow = r0 + rowBase + r8 + 8 * g;
      gi_g[brow * 192 + col] = c[r8] + bc;
    }
  }
}

// ---------------- kernel 2: 3 GRU+MLP iterations, slots resident in LDS ----
// Note: attn == 1.0 exactly in the reference (softmax normalized by itself over
// the size-1 token axis), so upd[b,s,:] = v[b,:] and gi is iteration-invariant.
__global__ __launch_bounds__(128) void iterate(
    const float* __restrict__ eps, const float* __restrict__ mu,
    const float* __restrict__ sigma, const float* __restrict__ gi_g,
    const _Float16* __restrict__ whh, const _Float16* __restrict__ w1,
    const _Float16* __restrict__ w2, const float* __restrict__ b_hh,
    const float* __restrict__ b1, const float* __restrict__ b2,
    const float* __restrict__ g_mlp, const float* __restrict__ b_mlp,
    float* __restrict__ slots_out) {
  __shared__ __align__(16) float    sl[64 * 64];     // 16 KB  current slots (2 batch x 32 slots)
  __shared__ __align__(16) float    giL[2 * 192];    // gi for the 2 batch elements
  __shared__ __align__(16) _Float16 hln[64 * 64];    // 8 KB   LN(slots) f16
  __shared__ __align__(16) _Float16 h16[64 * 128];   // 16 KB  gelu(hidden) f16

  const int tid = threadIdx.x;
  const int wave = tid >> 5, lane = tid & 31;
  const int g = lane >> 4, ln15 = lane & 15;
  const int rowBase = wave * 16;     // wave owns 16 slot-rows
  const int bb = wave >> 1;          // batch element within WG

  // slots = mu + sigma * eps  (rows = b*32+s, this WG covers 2 batches = 4096 elems)
  for (int i = tid; i < 64 * 64; i += 128)
    sl[i] = mu[i & 2047] + sigma[i & 2047] * eps[(size_t)blockIdx.x * 4096 + i];
  for (int i = tid; i < 2 * 192; i += 128)
    giL[i] = gi_g[(size_t)blockIdx.x * 2 * 192 + i];
  __syncthreads();

  // iteration-invariant per-lane gate constants
  float girc[4], gizc[4], ginc[4], bhr[4], bhz[4], bhn[4], b1v[8], b2v[4];
#pragma unroll
  for (int t = 0; t < 4; ++t) {
    int col = ln15 + 16 * t;
    girc[t] = giL[bb * 192 + col];
    gizc[t] = giL[bb * 192 + 64 + col];
    ginc[t] = giL[bb * 192 + 128 + col];
    bhr[t] = b_hh[col]; bhz[t] = b_hh[64 + col]; bhn[t] = b_hh[128 + col];
    b2v[t] = b2[col];
  }
#pragma unroll
  for (int nt = 0; nt < 8; ++nt) b1v[nt] = b1[nt * 16 + ln15];

  for (int it = 0; it < ITERS; ++it) {
    // ---- gh = slots @ W_hh  (K=64 -> 2 k-steps, N=192 -> 12 n-tiles) ----
    v16h a0 = ldsA_f32(sl + rowBase * 64, 64, 0, lane);
    v16h a1 = ldsA_f32(sl + rowBase * 64, 64, 32, lane);
    v8f acc[12];
#pragma unroll
    for (int nt = 0; nt < 12; ++nt) {
      v8f c = {};
      c = wmma16(a0, ldB(whh, nt * 2 + 0, lane), c);
      c = wmma16(a1, ldB(whh, nt * 2 + 1, lane), c);
      acc[nt] = c;
    }
    // ---- GRU gates (pure per-lane f32 VALU) ----
#pragma unroll
    for (int t = 0; t < 4; ++t) {
      int col = ln15 + 16 * t;
#pragma unroll
      for (int r8 = 0; r8 < 8; ++r8) {
        int row = rowBase + r8 + 8 * g;
        float prev = sl[row * 64 + col];
        float r = sigm(girc[t] + acc[t][r8] + bhr[t]);
        float z = sigm(gizc[t] + acc[t + 4][r8] + bhz[t]);
        float n = tanhf(ginc[t] + r * (acc[t + 8][r8] + bhn[t]));
        sl[row * 64 + col] = (1.f - z) * n + z * prev;
      }
    }
    __syncthreads();
    // ---- h = LN(slots) -> hln (f16) ----
    if (tid < 64) {
      const float* rp = sl + tid * 64;
      float m = 0.f;
      for (int d = 0; d < 64; ++d) m += rp[d];
      m *= (1.f / 64.f);
      float var = 0.f;
      for (int d = 0; d < 64; ++d) { float t = rp[d] - m; var += t * t; }
      var *= (1.f / 64.f);
      float inv = rsqrtf(var + LN_EPS);
      for (int d = 0; d < 64; ++d)
        hln[tid * 64 + d] = (_Float16)((rp[d] - m) * inv * g_mlp[d] + b_mlp[d]);
    }
    __syncthreads();
    // ---- H = gelu(h @ W1 + b1)  (N=128 -> 8 n-tiles) -> h16 ----
    v16h ha0 = ldsA_f16(hln + rowBase * 64, 64, 0, lane);
    v16h ha1 = ldsA_f16(hln + rowBase * 64, 64, 32, lane);
#pragma unroll
    for (int nt = 0; nt < 8; ++nt) {
      v8f c = {};
      c = wmma16(ha0, ldB(w1, nt * 2 + 0, lane), c);
      c = wmma16(ha1, ldB(w1, nt * 2 + 1, lane), c);
      int col = nt * 16 + ln15;
#pragma unroll
      for (int r8 = 0; r8 < 8; ++r8)
        h16[(rowBase + r8 + 8 * g) * 128 + col] = (_Float16)gelu_exact(c[r8] + b1v[nt]);
    }
    __syncthreads();
    // ---- slots += H @ W2 + b2  (K=128 -> 4 k-steps, N=64 -> 4 n-tiles) ----
    v16h ma[4];
#pragma unroll
    for (int kt = 0; kt < 4; ++kt)
      ma[kt] = ldsA_f16(h16 + rowBase * 128, 128, kt * 32, lane);
#pragma unroll
    for (int nt = 0; nt < 4; ++nt) {
      v8f c = {};
#pragma unroll
      for (int kt = 0; kt < 4; ++kt) c = wmma16(ma[kt], ldB(w2, nt * 4 + kt, lane), c);
      int col = nt * 16 + ln15;
#pragma unroll
      for (int r8 = 0; r8 < 8; ++r8) {
        int row = rowBase + r8 + 8 * g;
        sl[row * 64 + col] += c[r8] + b2v[nt];
      }
    }
    __syncthreads();
  }
  for (int i = tid; i < 64 * 64; i += 128)
    slots_out[(size_t)blockIdx.x * 4096 + i] = sl[i];
}

// ---------------- kernel 3: per-slot property heads ----------------
__global__ __launch_bounds__(256) void heads(
    const float* __restrict__ slots, const float* __restrict__ Wh1,
    const float* __restrict__ bh1, const float* __restrict__ Wh2,
    const float* __restrict__ bh2, float* __restrict__ pv) {
  int i = blockIdx.x * 256 + threadIdx.x;  // i = b*9 + k
  if (i >= BATCH * NUM_KNOWN) return;
  int b = i / NUM_KNOWN, k = i % NUM_KNOWN;
  const float* s = slots + ((size_t)b * NUM_SLOTS + k) * SLOT_DIM;
  float a[32];
#pragma unroll
  for (int j = 0; j < 32; ++j) a[j] = 0.f;
  for (int d = 0; d < 64; ++d) {
    float sv = s[d];
    const float* wrow = Wh1 + ((size_t)k * 64 + d) * 32;
#pragma unroll
    for (int j = 0; j < 32; ++j) a[j] += sv * wrow[j];
  }
  float acc2 = bh2[k];
#pragma unroll
  for (int j = 0; j < 32; ++j)
    acc2 += gelu_exact(a[j] + bh1[k * 32 + j]) * Wh2[k * 32 + j];
  pv[i] = sigm(acc2);
}

// ---------------- kernel 4: free-slot activation norms (deterministic) ------
__global__ __launch_bounds__(256) void free_act(const float* __restrict__ slots,
                                                float* __restrict__ out) {
  __shared__ float red[256];
  int s = NUM_KNOWN + blockIdx.x;  // 9..31
  float sum = 0.f;
  for (int b = threadIdx.x; b < BATCH; b += 256) {
    const float* p = slots + ((size_t)b * NUM_SLOTS + s) * SLOT_DIM;
    float q = 0.f;
#pragma unroll 8
    for (int d = 0; d < 64; ++d) q += p[d] * p[d];
    sum += sqrtf(q);
  }
  red[threadIdx.x] = sum;
  __syncthreads();
  for (int w = 128; w > 0; w >>= 1) {
    if (threadIdx.x < w) red[threadIdx.x] += red[threadIdx.x + w];
    __syncthreads();
  }
  if (threadIdx.x == 0) out[blockIdx.x] = red[0] * (1.f / (float)BATCH);
}

// ---------------- launcher ----------------
extern "C" void kernel_launch(void* const* d_in, const int* in_sizes, int n_in,
                              void* d_out, int out_size, void* d_ws, size_t ws_size,
                              hipStream_t stream) {
  const float* ws_in = (const float*)d_in[0];   // world_state
  const float* eps   = (const float*)d_in[1];
  const float* mu    = (const float*)d_in[2];
  const float* sigma = (const float*)d_in[3];
  const float* Wi  = (const float*)d_in[4];  const float* bi  = (const float*)d_in[5];
  // d_in[6..9] = Wk,bk,Wq,bq : mathematically dead (attn == 1 exactly), skipped.
  const float* Wv  = (const float*)d_in[10]; const float* bv  = (const float*)d_in[11];
  const float* Wih = (const float*)d_in[12]; const float* bih = (const float*)d_in[13];
  const float* Whh = (const float*)d_in[14]; const float* bhh = (const float*)d_in[15];
  const float* W1  = (const float*)d_in[16]; const float* b1  = (const float*)d_in[17];
  const float* W2  = (const float*)d_in[18]; const float* b2  = (const float*)d_in[19];
  const float* g_in  = (const float*)d_in[20]; const float* b_in  = (const float*)d_in[21];
  // d_in[22..23] = g_sl,b_sl : only feed q -> dead.
  const float* g_mlp = (const float*)d_in[24]; const float* b_mlp = (const float*)d_in[25];
  const float* Wh1 = (const float*)d_in[26]; const float* bh1 = (const float*)d_in[27];
  const float* Wh2 = (const float*)d_in[28]; const float* bh2 = (const float*)d_in[29];

  // outputs: pv (B*9) ++ slots (B*32*64) ++ free_act (23)
  float* pv    = (float*)d_out;
  float* slots = pv + (size_t)BATCH * NUM_KNOWN;
  float* freea = slots + (size_t)BATCH * NUM_SLOTS * SLOT_DIM;

  // workspace layout (needs ~12.7 MB): gi (B*192 f32) then f16-swizzled weights
  char* wsb = (char*)d_ws;
  float* gi = (float*)wsb;
  size_t off = (size_t)BATCH * 192 * sizeof(float);
  _Float16* wi_sw  = (_Float16*)(wsb + off); off += (size_t)256 * 64 * 2;
  _Float16* wv_sw  = (_Float16*)(wsb + off); off += (size_t)64 * 64 * 2;
  _Float16* wih_sw = (_Float16*)(wsb + off); off += (size_t)64 * 192 * 2;
  _Float16* whh_sw = (_Float16*)(wsb + off); off += (size_t)64 * 192 * 2;
  _Float16* w1_sw  = (_Float16*)(wsb + off); off += (size_t)64 * 128 * 2;
  _Float16* w2_sw  = (_Float16*)(wsb + off); off += (size_t)128 * 64 * 2;

  // kernel 0: weight swizzles
  swizzle_w<<<(256 * 64 + 255) / 256, 256, 0, stream>>>(Wi, wi_sw, 256, 64);
  swizzle_w<<<(64 * 64 + 255) / 256, 256, 0, stream>>>(Wv, wv_sw, 64, 64);
  swizzle_w<<<(64 * 192 + 255) / 256, 256, 0, stream>>>(Wih, wih_sw, 64, 192);
  swizzle_w<<<(64 * 192 + 255) / 256, 256, 0, stream>>>(Whh, whh_sw, 64, 192);
  swizzle_w<<<(64 * 128 + 255) / 256, 256, 0, stream>>>(W1, w1_sw, 64, 128);
  swizzle_w<<<(128 * 64 + 255) / 256, 256, 0, stream>>>(W2, w2_sw, 128, 64);

  // kernel 1: x/v/gi (64 batch rows per WG)
  preproc<<<BATCH / 64, 128, 0, stream>>>(ws_in, wi_sw, bi, wv_sw, bv, wih_sw, bih,
                                          g_in, b_in, gi);
  // kernel 2: 3 GRU+MLP iterations, 2 batch elements per WG
  iterate<<<BATCH / 2, 128, 0, stream>>>(eps, mu, sigma, gi, whh_sw, w1_sw, w2_sw,
                                         bhh, b1, b2, g_mlp, b_mlp, slots);
  // kernel 3: property heads
  heads<<<(BATCH * NUM_KNOWN + 255) / 256, 256, 0, stream>>>(slots, Wh1, bh1, Wh2, bh2, pv);
  // kernel 4: free slot norms
  free_act<<<NUM_SLOTS - NUM_KNOWN, 256, 0, stream>>>(slots, freea);
}
